// GLGXCA_30279519436938
// MI455X (gfx1250) — matrix operand order repaired and checked by reading, not compiled, not run on hardware
//
#include <hip/hip_runtime.h>
#include <hip/hip_bf16.h>
#include <math.h>
#include <stdint.h>

typedef __attribute__((ext_vector_type(16))) _Float16 v16h;
typedef __attribute__((ext_vector_type(8)))  _Float16 v8h;
typedef __attribute__((ext_vector_type(8)))  float    v8f;
typedef __attribute__((ext_vector_type(4))) unsigned int u32x4;
typedef __attribute__((ext_vector_type(8))) int          i32x8;
typedef __attribute__((ext_vector_type(4))) int          i32x4;

constexpr int Bz = 32, Cc = 512, Hh = 32, Ww = 32;
constexpr int Nn = Hh * Ww;          // 1024
constexpr int HEADS = 8, HD = 64;

// ---- workspace layout (bytes) ----
constexpr size_t SZ_XT  = (size_t)Bz * Nn * Cc * 2;          // x_flat f16 [B,N,C]   32MB
constexpr size_t SZ_W16 = (size_t)4 * Cc * Cc * 2;           // Wq,Wk,Wv,Wp f16       2MB
constexpr size_t SZ_P16 = (size_t)Bz * Cc * Nn * 2;          // one projection f16   32MB
constexpr size_t SZ_G   = (size_t)Bz * Cc * Nn * 4;          // dwconv out f32       64MB
constexpr size_t SZ_ATT = (size_t)Bz * HEADS * HD * HD * 4;  // attn f32              4MB

constexpr size_t OFF_XT    = 0;
constexpr size_t OFF_W16   = OFF_XT   + SZ_XT;
constexpr size_t OFF_Q16   = OFF_W16  + SZ_W16;
constexpr size_t OFF_K16   = OFF_Q16  + SZ_P16;
constexpr size_t OFF_V16   = OFF_K16  + SZ_P16;
constexpr size_t OFF_G     = OFF_V16  + SZ_P16;
constexpr size_t OFF_ATTN  = OFF_G    + SZ_G;
constexpr size_t OFF_A16   = OFF_ATTN + SZ_ATT;
constexpr size_t OFF_O16   = OFF_A16  + SZ_ATT / 2;
constexpr size_t OFF_STATS = OFF_O16  + SZ_P16;

#define DEV __device__ __forceinline__

DEV v16h pack16(v8h lo, v8h hi) {
  v16h r;
#pragma unroll
  for (int i = 0; i < 8; ++i) { r[i] = lo[i]; r[8 + i] = hi[i]; }
  return r;
}

// A fragment (16x32 f16, row-major source, row stride lda):
// lane<16: row m, K {0..7} and {16..23}; lane>=16: K shifted by +8.
DEV v16h frag_a(const _Float16* A, int lda, int row, int k0, int lane) {
  const int hf = lane >> 4;
  const _Float16* p = A + (size_t)row * lda + k0 + hf * 8;
  return pack16(*(const v8h*)p, *(const v8h*)(p + 16));
}

// B fragment (32x16 f16): column col of the B-matrix is a contiguous K-run in
// memory (source stored [col][k], row stride ldb). lane<16: K {0..7} and {16..23},
// lane>=16: K shifted by +8.
DEV v16h frag_b(const _Float16* Bc, int ldb, int col, int k0, int lane) {
  const int hf = lane >> 4;
  const _Float16* p = Bc + (size_t)col * ldb + k0 + hf * 8;
  return pack16(*(const v8h*)p, *(const v8h*)(p + 16));
}

// ---- Tensor Data Mover: stage a rows x 32 f16 tile (row stride `stride0`
// elements, tensor width Cc) from global into LDS. Issued by one wave;
// completion tracked with TENSORcnt. D# layout per CDNA5 ISA 08_async_tensor.
// 6-arg builtin form (upstream clang-23 / therock-10.0 headers).
DEV void stage_tile(_Float16* ldsPtr, const _Float16* gaddr, int rows,
                    int tensor_d1, int stride0) {
#if __has_builtin(__builtin_amdgcn_tensor_load_to_lds)
  u32x4 g0 = {};
  i32x8 g1 = {};
  i32x4 g2 = {}, g3 = {};
  i32x8 g4 = {};
  const unsigned long long ga = (unsigned long long)(uintptr_t)gaddr;
  g0[0] = 1u;                                         // count=1 (valid user D#)
  g0[1] = (unsigned)(uintptr_t)ldsPtr;                // lds_addr = flat[31:0]
  g0[2] = (unsigned)ga;                               // global_addr[31:0]
  g0[3] = (unsigned)((ga >> 32) & 0x01ffffffu) | (2u << 30);  // addr[56:32], type=2
  g1[0] = 1 << 16;                                    // data_size = 2 bytes
  g1[1] = (Cc & 0xffff) << 16;                        // tensor_dim0[15:0] @ 63:48
  g1[2] = (int)(((unsigned)Cc >> 16) | (((unsigned)tensor_d1 & 0xffffu) << 16));
  g1[3] = (int)(((unsigned)tensor_d1 >> 16) | (32u << 16));   // tile_dim0 = 32
  g1[4] = rows;                                       // tile_dim1 (tile_dim2=0)
  g1[5] = stride0;                                    // tensor_dim0_stride[31:0]
  g1[6] = 0;
  g1[7] = 0;
  __builtin_amdgcn_tensor_load_to_lds(g0, g1, g2, g3, g4, 0);
#else
  const int lane = threadIdx.x & 31;                  // fallback: wave copy
  for (int e = lane; e < rows * 32; e += 32)
    ldsPtr[e] = gaddr[(size_t)(e >> 5) * stride0 + (e & 31)];
#endif
}

DEV void wait_tdm() {
#if __has_builtin(__builtin_amdgcn_s_wait_tensorcnt)
  __builtin_amdgcn_s_wait_tensorcnt(0);
#endif
}

// ---------- prep ----------
__global__ __launch_bounds__(256) void k_cvt_w(const float* __restrict__ Wq,
                                               const float* __restrict__ Wk,
                                               const float* __restrict__ Wv,
                                               const float* __restrict__ Wp,
                                               _Float16* __restrict__ w16) {
  const int idx = blockIdx.x * 256 + threadIdx.x;          // 4*C*C
  const int which = idx >> 18;                             // C*C = 2^18
  const int r = idx & (Cc * Cc - 1);
  const float* src = (which == 0) ? Wq : (which == 1) ? Wk : (which == 2) ? Wv : Wp;
  w16[idx] = (_Float16)src[r];
}

__global__ __launch_bounds__(256) void k_xT(const float* __restrict__ x,
                                            _Float16* __restrict__ xt) {
  const int idx = blockIdx.x * 256 + threadIdx.x;          // (b*N+n)*C + c
  const int c = idx & (Cc - 1);
  const int n = (idx >> 9) & (Nn - 1);
  const int b = idx >> 19;
  xt[idx] = (_Float16)x[((size_t)b * Cc + c) * Nn + n];
}

// ---------- gate path: dwconv3x3 + batch stats ----------
__global__ __launch_bounds__(256) void k_dwconv(const float* __restrict__ x,
                                                const float* __restrict__ dw,
                                                float* __restrict__ g,
                                                float* __restrict__ sum,
                                                float* __restrict__ sumsq) {
  __shared__ float r1[256], r2[256];
  const int p = blockIdx.x, b = p / Cc, c = p % Cc;
  const int t = threadIdx.x;
  float wgt[9];
#pragma unroll
  for (int i = 0; i < 9; ++i) wgt[i] = dw[c * 9 + i];
  const float* in = x + ((size_t)b * Cc + c) * Nn;
  float* go = g + ((size_t)b * Cc + c) * Nn;
  float s1 = 0.f, s2 = 0.f;
#pragma unroll
  for (int i = 0; i < 4; ++i) {
    const int pix = t + i * 256;
    const int y = pix >> 5, xx = pix & 31;
    float a = 0.f;
#pragma unroll
    for (int ky = 0; ky < 3; ++ky) {
      const int iy = y + ky - 1;
      if (iy < 0 || iy >= Hh) continue;
#pragma unroll
      for (int kx = 0; kx < 3; ++kx) {
        const int ix = xx + kx - 1;
        if (ix < 0 || ix >= Ww) continue;
        a += wgt[ky * 3 + kx] * in[iy * Ww + ix];
      }
    }
    go[pix] = a; s1 += a; s2 += a * a;
  }
  r1[t] = s1; r2[t] = s2; __syncthreads();
  for (int o = 128; o > 0; o >>= 1) {
    if (t < o) { r1[t] += r1[t + o]; r2[t] += r2[t + o]; }
    __syncthreads();
  }
  if (t == 0) { atomicAdd(&sum[c], r1[0]); atomicAdd(&sumsq[c], r2[0]); }
}

__global__ __launch_bounds__(256) void k_bn_final(const float* __restrict__ sum,
                                                  const float* __restrict__ sumsq,
                                                  const float* __restrict__ gamma,
                                                  const float* __restrict__ beta,
                                                  float* __restrict__ scale,
                                                  float* __restrict__ shift) {
  const int c = blockIdx.x * 256 + threadIdx.x;
  if (c >= Cc) return;
  const float cnt = (float)(Bz * Nn);
  const float mean = sum[c] / cnt;
  const float var = sumsq[c] / cnt - mean * mean;
  const float sc = gamma[c] * rsqrtf(var + 1e-5f);
  scale[c] = sc;
  shift[c] = beta[c] - mean * sc;
}

// =====================================================================
// Macro-tiled projection GEMM: block = 256 thr (8 waves) computes 128x64.
// K staged in 32-wide slices via TDM into double-buffered LDS; wave w owns
// rows [w*16, w*16+16) and all 4 column tiles (A-fragment reuse x4).
// =====================================================================
#define PROJ_PROLOG(APTR, WPTR)                                               \
  __shared__ _Float16 As[2][128 * 32];                                        \
  __shared__ _Float16 Bs[2][64 * 32];                                         \
  const int wid = threadIdx.x >> 5, lane = threadIdx.x & 31;                  \
  const int lm = lane & 15, hf = lane >> 4;                                   \
  const int rowblk = blockIdx.x >> 3, colblk = blockIdx.x & 7;                \
  const int m0g = rowblk * 128, n0g = colblk * 64;                            \
  v8f acc[4] = {{}, {}, {}, {}};                                              \
  const bool issuer = (wid == 0);                                             \
  if (issuer) {                                                               \
    stage_tile(As[0], (APTR) + (size_t)m0g * Cc, 128, Bz * Nn, Cc);           \
    stage_tile(Bs[0], (WPTR) + (size_t)n0g * Cc, 64, Cc, Cc);                 \
  }                                                                           \
  for (int i = 0; i < 16; ++i) {                                              \
    if (issuer) wait_tdm();                                                   \
    __syncthreads();                                                          \
    if (issuer && i + 1 < 16) {                                               \
      const int k0 = (i + 1) * 32, buf = (i + 1) & 1;                         \
      stage_tile(As[buf], (APTR) + (size_t)m0g * Cc + k0, 128, Bz * Nn, Cc);  \
      stage_tile(Bs[buf], (WPTR) + (size_t)n0g * Cc + k0, 64, Cc, Cc);        \
    }                                                                         \
    const _Float16* Ab = As[i & 1];                                           \
    const _Float16* Bb = Bs[i & 1];                                           \
    v16h afr = frag_a(Ab, 32, wid * 16 + lm, 0, lane);                        \
    _Pragma("unroll")                                                         \
    for (int tt = 0; tt < 4; ++tt) {                                          \
      v16h bfr = frag_b(Bb, 32, tt * 16 + lm, 0, lane);                       \
      acc[tt] = __builtin_amdgcn_wmma_f32_16x16x32_f16(                       \
          false, afr, false, bfr, (short)0, acc[tt], false, false);           \
    }                                                                         \
  }

// ---------- q/k projection -> [B,H,HD,N] f16 ----------
__global__ __launch_bounds__(256) void k_proj_qk(const _Float16* __restrict__ xt,
                                                 const _Float16* __restrict__ w16,
                                                 _Float16* __restrict__ q16,
                                                 _Float16* __restrict__ k16) {
  const _Float16* W = w16 + (size_t)blockIdx.y * Cc * Cc;
  _Float16* dst = blockIdx.y ? k16 : q16;
  PROJ_PROLOG(xt, W)
#pragma unroll
  for (int tt = 0; tt < 4; ++tt) {
    const int co = n0g + tt * 16 + lm, h = co >> 6, d = co & 63;
#pragma unroll
    for (int j = 0; j < 8; ++j) {
      const int m = m0g + wid * 16 + hf * 8 + j;
      const int b_ = m >> 10, n = m & (Nn - 1);
      dst[(((size_t)b_ * HEADS + h) * HD + d) * Nn + n] = (_Float16)acc[tt][j];
    }
  }
}

// ---------- v projection with fused BN+SiLU gate -> [B,H,N,HD] f16 ----------
__global__ __launch_bounds__(256) void k_proj_v(const _Float16* __restrict__ xt,
                                                const _Float16* __restrict__ wv16,
                                                const float* __restrict__ g,
                                                const float* __restrict__ scale,
                                                const float* __restrict__ shift,
                                                _Float16* __restrict__ v16) {
  PROJ_PROLOG(xt, wv16)
#pragma unroll
  for (int tt = 0; tt < 4; ++tt) {
    const int co = n0g + tt * 16 + lm, h = co >> 6, e = co & 63;
    const float sc = scale[co], sh = shift[co];
#pragma unroll
    for (int j = 0; j < 8; ++j) {
      const int m = m0g + wid * 16 + hf * 8 + j;
      const int b_ = m >> 10, n = m & (Nn - 1);
      float gv = g[((size_t)b_ * Cc + co) * Nn + n] * sc + sh;
      const float silu = gv / (1.f + expf(-gv));
      v16[(((size_t)b_ * HEADS + h) * Nn + n) * HD + e] = (_Float16)(acc[tt][j] * silu);
    }
  }
}

// ---------- output projection + bias -> fp32 NCHW ----------
__global__ __launch_bounds__(256) void k_proj_out(const _Float16* __restrict__ o16,
                                                  const _Float16* __restrict__ wp16,
                                                  const float* __restrict__ bp,
                                                  float* __restrict__ out) {
  PROJ_PROLOG(o16, wp16)
#pragma unroll
  for (int tt = 0; tt < 4; ++tt) {
    const int co = n0g + tt * 16 + lm;
    const float bias = bp[co];
#pragma unroll
    for (int j = 0; j < 8; ++j) {
      const int m = m0g + wid * 16 + hf * 8 + j;
      const int b_ = m >> 10, n = m & (Nn - 1);
      out[((size_t)b_ * Cc + co) * Nn + n] = acc[tt][j] + bias;
    }
  }
}

// ---------- in-place L2 norm over N for q16 and k16 ----------
__global__ __launch_bounds__(256) void k_l2norm(_Float16* __restrict__ q16,
                                                _Float16* __restrict__ k16) {
  __shared__ float red[256];
  const int row = blockIdx.x;                    // 2 * B*H*HD rows
  const int R = Bz * HEADS * HD;
  _Float16* p = (row < R) ? (q16 + (size_t)row * Nn) : (k16 + (size_t)(row - R) * Nn);
  const int t = threadIdx.x;
  float s = 0.f;
#pragma unroll
  for (int i = 0; i < 4; ++i) { float v = (float)p[t + i * 256]; s += v * v; }
  red[t] = s; __syncthreads();
  for (int o = 128; o > 0; o >>= 1) { if (t < o) red[t] += red[t + o]; __syncthreads(); }
  const float inv = 1.f / fmaxf(sqrtf(red[0]), 1e-12f);
#pragma unroll
  for (int i = 0; i < 4; ++i) {
    const int idx = t + i * 256;
    p[idx] = (_Float16)((float)p[idx] * inv);
  }
}

// ---------- attn = q^T k per head (M=N=64, K=1024), * temperature ----------
__global__ __launch_bounds__(256) void k_attn(const _Float16* __restrict__ q16,
                                              const _Float16* __restrict__ k16,
                                              const float* __restrict__ temp,
                                              float* __restrict__ attnf) {
  const int lane = threadIdx.x & 31, lm = lane & 15, hf = lane >> 4;
  const int tile = blockIdx.x * 8 + (threadIdx.x >> 5);    // 4096
  const int bh = tile >> 4, t4 = tile & 15;
  const int m0 = (t4 >> 2) * 16, n0 = (t4 & 3) * 16;
  const _Float16* A = q16 + (size_t)bh * HD * Nn;          // [d][n]
  const _Float16* Bc = k16 + (size_t)bh * HD * Nn;         // k rows are B columns
  v8f acc = {};
#pragma unroll 4
  for (int k0 = 0; k0 < Nn; k0 += 32) {
    v16h a = frag_a(A, Nn, m0 + lm, k0, lane);
    v16h b = frag_b(Bc, Nn, n0 + lm, k0, lane);
    acc = __builtin_amdgcn_wmma_f32_16x16x32_f16(false, a, false, b, (short)0, acc, false, false);
  }
  const float tp = temp[bh & (HEADS - 1)];
  float* O = attnf + (size_t)bh * HD * HD;
#pragma unroll
  for (int j = 0; j < 8; ++j)
    O[(size_t)(m0 + hf * 8 + j) * HD + n0 + lm] = acc[j] * tp;
}

// ---------- softmax over last dim (rows of 64), one wave per row ----------
__global__ __launch_bounds__(256) void k_softmax(const float* __restrict__ attnf,
                                                 _Float16* __restrict__ attn16) {
  const int lane = threadIdx.x & 31;
  const int row = blockIdx.x * 8 + (threadIdx.x >> 5);     // 16384 rows
  const float* r = attnf + (size_t)row * HD;
  const float a0 = r[lane], a1 = r[lane + 32];
  float m = fmaxf(a0, a1);
#pragma unroll
  for (int o = 16; o > 0; o >>= 1) m = fmaxf(m, __shfl_xor(m, o, 32));
  const float e0 = expf(a0 - m), e1 = expf(a1 - m);
  float s = e0 + e1;
#pragma unroll
  for (int o = 16; o > 0; o >>= 1) s += __shfl_xor(s, o, 32);
  const float inv = 1.f / s;
  _Float16* w = attn16 + (size_t)row * HD;
  w[lane] = (_Float16)(e0 * inv);
  w[lane + 32] = (_Float16)(e1 * inv);
}

// ---------- out_head = attn @ v (M=64, N=1024, K=64) -> o16 [B,N,C] ----------
__global__ __launch_bounds__(256) void k_attnv(const _Float16* __restrict__ attn16,
                                               const _Float16* __restrict__ v16,
                                               _Float16* __restrict__ o16) {
  const int lane = threadIdx.x & 31, lm = lane & 15, hf = lane >> 4;
  const int tile = blockIdx.x * 8 + (threadIdx.x >> 5);    // 65536
  const int bh = tile >> 8, t8 = tile & 255;
  const int m0 = (t8 >> 6) * 16, n0 = (t8 & 63) * 16;
  const _Float16* A = attn16 + (size_t)bh * HD * HD;       // [d][e], lda=64
  const _Float16* Bc = v16 + (size_t)bh * Nn * HD;         // [n][e], cols contiguous in e
  v8f acc = {};
#pragma unroll
  for (int k0 = 0; k0 < HD; k0 += 32) {
    v16h a = frag_a(A, HD, m0 + lm, k0, lane);
    v16h b = frag_b(Bc, HD, n0 + lm, k0, lane);
    acc = __builtin_amdgcn_wmma_f32_16x16x32_f16(false, a, false, b, (short)0, acc, false, false);
  }
  const int b_ = bh >> 3, h = bh & (HEADS - 1);
  const int ncol = n0 + lm;
#pragma unroll
  for (int j = 0; j < 8; ++j) {
    const int d = m0 + hf * 8 + j;
    o16[((size_t)b_ * Nn + ncol) * Cc + h * HD + d] = (_Float16)acc[j];
  }
}

extern "C" void kernel_launch(void* const* d_in, const int* in_sizes, int n_in,
                              void* d_out, int out_size, void* d_ws, size_t ws_size,
                              hipStream_t stream) {
  const float* x    = (const float*)d_in[0];
  const float* Wq   = (const float*)d_in[1];
  const float* Wk   = (const float*)d_in[2];
  const float* Wv   = (const float*)d_in[3];
  const float* dw   = (const float*)d_in[4];
  const float* gam  = (const float*)d_in[5];
  const float* bet  = (const float*)d_in[6];
  const float* temp = (const float*)d_in[7];
  const float* Wp   = (const float*)d_in[8];
  const float* bp   = (const float*)d_in[9];
  float* out = (float*)d_out;

  char* w = (char*)d_ws;
  _Float16* xt     = (_Float16*)(w + OFF_XT);
  _Float16* w16    = (_Float16*)(w + OFF_W16);
  _Float16* q16    = (_Float16*)(w + OFF_Q16);
  _Float16* k16    = (_Float16*)(w + OFF_K16);
  _Float16* v16    = (_Float16*)(w + OFF_V16);
  float*    g      = (float*)(w + OFF_G);
  float*    attnf  = (float*)(w + OFF_ATTN);
  _Float16* attn16 = (_Float16*)(w + OFF_A16);
  _Float16* o16    = (_Float16*)(w + OFF_O16);
  float*    csum   = (float*)(w + OFF_STATS);
  float*    csq    = csum + Cc;
  float*    scal   = csum + 2 * Cc;
  float*    shft   = csum + 3 * Cc;

  (void)hipMemsetAsync(csum, 0, 2 * Cc * sizeof(float), stream);

  k_cvt_w <<<(4 * Cc * Cc) / 256, 256, 0, stream>>>(Wq, Wk, Wv, Wp, w16);
  k_xT    <<<(Bz * Nn * Cc) / 256, 256, 0, stream>>>(x, xt);
  k_dwconv<<<Bz * Cc, 256, 0, stream>>>(x, dw, g, csum, csq);
  k_bn_final<<<(Cc + 255) / 256, 256, 0, stream>>>(csum, csq, gam, bet, scal, shft);

  k_proj_qk<<<dim3(2048, 2), 256, 0, stream>>>(xt, w16, q16, k16);
  k_l2norm<<<2 * Bz * HEADS * HD, 256, 0, stream>>>(q16, k16);
  k_proj_v<<<2048, 256, 0, stream>>>(xt, w16 + 2 * (size_t)Cc * Cc, g, scal, shft, v16);

  k_attn   <<<512, 256, 0, stream>>>(q16, k16, temp, attnf);
  k_softmax<<<2048, 256, 0, stream>>>(attnf, attn16);
  k_attnv  <<<8192, 256, 0, stream>>>(attn16, v16, o16);
  k_proj_out<<<2048, 256, 0, stream>>>(o16, w16 + 3 * (size_t)Cc * Cc, bp, out);

  (void)in_sizes; (void)n_in; (void)out_size; (void)ws_size;
}
